// GINNet_36137854829159
// MI455X (gfx1250) — compile-verified, matrix operand order checked by reading, not run on hardware
//
#include <hip/hip_runtime.h>

#define D 128
#define LAYERS 5

typedef __attribute__((ext_vector_type(16))) __bf16 v16bf;
typedef __attribute__((ext_vector_type(8)))  __bf16 v8bf;
typedef __attribute__((ext_vector_type(8)))  float  v8f;

union V16U { v16bf v; struct { v8bf lo, hi; } p; };

static __device__ __forceinline__ unsigned short f32_to_bf16u(float f) {
    __bf16 b = (__bf16)f;
    return __builtin_bit_cast(unsigned short, b);
}

static __device__ __forceinline__ unsigned int pack_bf16x2(float a, float b) {
    return (unsigned int)f32_to_bf16u(a) | ((unsigned int)f32_to_bf16u(b) << 16);
}

// ---------------- simple streaming kernels ----------------

__global__ void k_f32_to_bf16(const float* __restrict__ in,
                              unsigned short* __restrict__ out, int n) {
    int i = blockIdx.x * blockDim.x + threadIdx.x;
    if (i < n) out[i] = f32_to_bf16u(in[i]);
}

__global__ void k_zero(float* __restrict__ p, int n) {
    int i = blockIdx.x * blockDim.x + threadIdx.x;
    if (i < n) p[i] = 0.0f;
}

__global__ void k_copy4(const float4* __restrict__ in, float4* __restrict__ out, int n4) {
    int i = blockIdx.x * blockDim.x + threadIdx.x;
    if (i < n4) out[i] = in[i];
}

// ---------------- edge scatter-add (dominant cost, L2-resident atomics) ----------------
// 32 threads per edge, float4 per thread: agg[dst] += h[src]
__global__ void k_edge_scatter(const float* __restrict__ h,
                               const int* __restrict__ ei,   // [2,E]
                               float* __restrict__ agg, long long E) {
    long long tid = (long long)blockIdx.x * blockDim.x + threadIdx.x;
    if (tid >= E * 32) return;
    long long e = tid >> 5;
    int c = (int)(tid & 31) * 4;
    int src = ei[e];
    int dst = ei[E + e];
    const float4 v = *(const float4*)(h + (long long)src * D + c);
    float* o = agg + (long long)dst * D + c;
    atomicAdd(o + 0, v.x);
    atomicAdd(o + 1, v.y);
    atomicAdd(o + 2, v.z);
    atomicAdd(o + 3, v.w);
}

// ---------------- WMMA GEMM: Out[M,128] = act(A[M,128] @ W^T + bias) ----------------
// W is bf16 row-major [128,128], row j = output feature j (B column j = W row j,
// contiguous along K -> clean b128 fragment loads).
// Block = 256 threads = 8 waves; block owns a 16-row strip, wave w owns cols [16w,16w+16).
// A tile staged in LDS as bf16, row stride 136 halves (16B pad) to spread banks.
// M must be a multiple of 16 (N=50000, G=2048 both are) -> branch-free epilogue.
template<bool A_BF16, bool RELU, bool OUT_BF16>
__global__ __launch_bounds__(256) void k_gemm128(
    const void* __restrict__ A,
    const unsigned short* __restrict__ W,
    const float* __restrict__ bias,
    void* __restrict__ Out, int M)
{
    __shared__ __attribute__((aligned(16))) unsigned short sA[16 * 136];

    const int m0 = blockIdx.x * 16;
    const int t  = threadIdx.x;

    // cooperative stage of A tile (16x128) into LDS as bf16:
    // each thread handles 8 contiguous elements = one b128 global load + one ds_store_b128
    {
        const int idx = t * 8;            // 0..2040, covers 16*128 exactly
        const int r = idx >> 7, c = idx & 127;
        uint4 v;
        if (A_BF16) {
            v = *(const uint4*)((const unsigned short*)A + (long long)(m0 + r) * D + c);
        } else {
            const float* ap = (const float*)A + (long long)(m0 + r) * D + c;
            const float4 f0 = *(const float4*)ap;
            const float4 f1 = *(const float4*)(ap + 4);
            v.x = pack_bf16x2(f0.x, f0.y);
            v.y = pack_bf16x2(f0.z, f0.w);
            v.z = pack_bf16x2(f1.x, f1.y);
            v.w = pack_bf16x2(f1.z, f1.w);
        }
        *(uint4*)(sA + r * 136 + c) = v;
    }
    __syncthreads();

    const int lane = t & 31;
    const int wave = t >> 5;
    const int n0   = wave * 16;
    const int mr   = lane & 15;            // A row in tile / B column
    const int khi  = (lane >> 4) * 8;      // per-half-wave K sub-offset (ISA 16-bit layout)

    // accumulator init = bias (same for all 8 row-VGPRs of this lane's column)
    const float bv0 = bias[n0 + mr];
    v8f acc;
#pragma unroll
    for (int r = 0; r < 8; ++r) acc[r] = bv0;

    const unsigned short* arow = sA + mr * 136;
    const unsigned short* wrow = W + (long long)(n0 + mr) * D;

#pragma unroll
    for (int k0 = 0; k0 < D; k0 += 32) {
        V16U a, b;
        a.p.lo = *(const v8bf*)(const void*)(arow + k0 + khi);
        a.p.hi = *(const v8bf*)(const void*)(arow + k0 + khi + 16);
        b.p.lo = *(const v8bf*)(const void*)(wrow + k0 + khi);
        b.p.hi = *(const v8bf*)(const void*)(wrow + k0 + khi + 16);
        acc = __builtin_amdgcn_wmma_f32_16x16x32_bf16(
            false, a.v, false, b.v, (short)0, acc, false, false);
    }

    // C layout: VGPR r -> row m0 + r + 8*(lane>=16), col n0 + (lane&15).
    // One 64-bit base per lane; the 8 stores use immediate offsets (r*D elements).
    const int rowbase = m0 + ((lane >> 4) << 3);
    const long long off0 = (long long)rowbase * D + n0 + mr;
    if (OUT_BF16) {
        unsigned short* o = (unsigned short*)Out + off0;
#pragma unroll
        for (int r = 0; r < 8; ++r) {
            float v = acc[r];
            if (RELU) v = fmaxf(v, 0.0f);
            o[r * D] = f32_to_bf16u(v);
        }
    } else {
        float* o = (float*)Out + off0;
#pragma unroll
        for (int r = 0; r < 8; ++r) {
            float v = acc[r];
            if (RELU) v = fmaxf(v, 0.0f);
            o[r * D] = v;
        }
    }
}

// ---------------- BatchNorm: per-column sum / sumsq partials ----------------
__global__ void k_bn_stats(const float* __restrict__ z, float* __restrict__ stats,
                           int M, int rowsPerBlock) {
    int col = threadIdx.x;                 // blockDim = 128
    int r0 = blockIdx.x * rowsPerBlock;
    int r1 = r0 + rowsPerBlock; if (r1 > M) r1 = M;
    float s = 0.0f, s2 = 0.0f;
    for (int r = r0; r < r1; ++r) {
        float v = z[(long long)r * D + col];
        s += v; s2 += v * v;
    }
    atomicAdd(&stats[col], s);
    atomicAdd(&stats[D + col], s2);
}

__global__ void k_bn_apply(const float* __restrict__ z, const float* __restrict__ stats,
                           const float* __restrict__ gamma, const float* __restrict__ beta,
                           float* __restrict__ h, int n, float invN) {
    int i = blockIdx.x * blockDim.x + threadIdx.x;
    if (i >= n) return;
    int col = i & (D - 1);
    float mu  = stats[col] * invN;
    float var = stats[D + col] * invN - mu * mu;
    float v = (z[i] - mu) * rsqrtf(var + 1e-5f) * gamma[col] + beta[col];
    h[i] = v > 0.0f ? v : 0.0f;
}

// ---------------- graph pooling (segment_sum via atomics, pooled is L2-resident) ----------------
__global__ void k_pool(const float* __restrict__ h, const int* __restrict__ batch,
                       float* __restrict__ pooled, int N) {
    long long tid = (long long)blockIdx.x * blockDim.x + threadIdx.x;
    if (tid >= (long long)N * 32) return;
    int n = (int)(tid >> 5);
    int c = (int)(tid & 31) * 4;
    int g = batch[n];
    const float4 v = *(const float4*)(h + (long long)n * D + c);
    float* o = pooled + (long long)g * D + c;
    atomicAdd(o + 0, v.x);
    atomicAdd(o + 1, v.y);
    atomicAdd(o + 2, v.z);
    atomicAdd(o + 3, v.w);
}

// ---------------- final [G,128] x [128] dot: wave per graph, shuffle reduce ----------------
__global__ void k_dot_out(const float* __restrict__ g1, const float* __restrict__ w2,
                          const float* __restrict__ b2, float* __restrict__ out, int G) {
    int gid  = (int)((blockIdx.x * blockDim.x + threadIdx.x) >> 5);
    int lane = threadIdx.x & 31;
    if (gid >= G) return;
    const float4 x = *(const float4*)(g1 + (long long)gid * D + lane * 4);
    const float4 w = *(const float4*)(w2 + lane * 4);
    float s = x.x * w.x + x.y * w.y + x.z * w.z + x.w * w.w;
#pragma unroll
    for (int off = 16; off; off >>= 1) s += __shfl_xor(s, off, 32);
    if (lane == 0) out[gid] = s + b2[0];
}

// ---------------- host orchestration ----------------
extern "C" void kernel_launch(void* const* d_in, const int* in_sizes, int n_in,
                              void* d_out, int out_size, void* d_ws, size_t ws_size,
                              hipStream_t stream) {
    const float* x       = (const float*)d_in[0];
    const int*   ei      = (const int*)d_in[1];
    const int*   batch   = (const int*)d_in[2];
    const float* lin_w   = (const float*)d_in[3];
    const float* lin_b   = (const float*)d_in[4];
    const float* conv_w1 = (const float*)d_in[5];
    const float* conv_b1 = (const float*)d_in[6];
    const float* conv_w2 = (const float*)d_in[7];
    const float* conv_b2 = (const float*)d_in[8];
    const float* bn_g    = (const float*)d_in[9];
    const float* bn_b    = (const float*)d_in[10];
    const float* mlp_w1  = (const float*)d_in[11];
    const float* mlp_b1  = (const float*)d_in[12];
    const float* mlp_w2  = (const float*)d_in[13];
    const float* mlp_b2  = (const float*)d_in[14];

    const int       N = in_sizes[0] / D;        // 50000
    const long long E = in_sizes[1] / 2;        // 1,600,000
    const int       G = out_size;               // 2048
    const int ND = N * D;

    // workspace carve-out (~67 MB)
    char* p = (char*)d_ws;
    float*          h    = (float*)p;          p += (size_t)ND * 4;
    float*          agg  = (float*)p;          p += (size_t)ND * 4;
    unsigned short* t1b  = (unsigned short*)p; p += (size_t)ND * 2;
    unsigned short* wbf  = (unsigned short*)p; p += (size_t)12 * D * D * 2;
    float*          pooled = (float*)p;        p += (size_t)G * D * 4;
    float*          gbuf = (float*)p;          p += (size_t)G * D * 4;
    float*          stats = (float*)p;         p += 2 * D * 4;

    unsigned short* lin_wb = wbf;
    unsigned short* c1b    = wbf + D * D;
    unsigned short* c2b    = c1b + LAYERS * D * D;
    unsigned short* m1b    = c2b + LAYERS * D * D;

    const int rowTiles = N / 16;                // 3125 (exact)
    const int gTiles   = G / 16;                // 128 (exact)
    const unsigned eBlocks = (unsigned)((E * 32 + 255) / 256);
    const unsigned pBlocks = (unsigned)(((long long)N * 32 + 255) / 256);

    // --- weights -> bf16 (once per launch; tiny) ---
    k_f32_to_bf16<<<(D*D + 255) / 256, 256, 0, stream>>>(lin_w, lin_wb, D * D);
    k_f32_to_bf16<<<(LAYERS*D*D + 255) / 256, 256, 0, stream>>>(conv_w1, c1b, LAYERS * D * D);
    k_f32_to_bf16<<<(LAYERS*D*D + 255) / 256, 256, 0, stream>>>(conv_w2, c2b, LAYERS * D * D);
    k_f32_to_bf16<<<(D*D + 255) / 256, 256, 0, stream>>>(mlp_w1, m1b, D * D);

    // --- input linear: h = x @ lin_w^T + lin_b ---
    k_gemm128<false, false, false><<<rowTiles, 256, 0, stream>>>(x, lin_wb, lin_b, h, N);

    for (int l = 0; l < LAYERS; ++l) {
        // agg = h  (folds the GIN "x + sum" into the scatter target)
        k_copy4<<<(ND/4 + 255) / 256, 256, 0, stream>>>((const float4*)h, (float4*)agg, ND / 4);
        // agg[dst] += h[src] over all edges
        k_edge_scatter<<<eBlocks, 256, 0, stream>>>(h, ei, agg, E);
        // t1b = relu(agg @ W1^T + b1)  (bf16 out, halves traffic for next GEMM)
        k_gemm128<false, true, true><<<rowTiles, 256, 0, stream>>>(
            agg, c1b + l * D * D, conv_b1 + l * D, t1b, N);
        // agg = t1b @ W2^T + b2  (fp32 for BN stats precision; agg is dead, reuse it)
        k_gemm128<true, false, false><<<rowTiles, 256, 0, stream>>>(
            t1b, c2b + l * D * D, conv_b2 + l * D, agg, N);
        // BatchNorm (training-mode, biased var) + ReLU -> h
        k_zero<<<1, 256, 0, stream>>>(stats, 2 * D);
        k_bn_stats<<<(N + 255) / 256, D, 0, stream>>>(agg, stats, N, 256);
        k_bn_apply<<<(ND + 255) / 256, 256, 0, stream>>>(
            agg, stats, bn_g + l * D, bn_b + l * D, h, ND, 1.0f / (float)N);
    }

    // --- readout ---
    k_zero<<<(G * D + 255) / 256, 256, 0, stream>>>(pooled, G * D);
    k_pool<<<pBlocks, 256, 0, stream>>>(h, batch, pooled, N);
    k_gemm128<false, true, false><<<gTiles, 256, 0, stream>>>(pooled, m1b, mlp_b1, gbuf, G);
    k_dot_out<<<(G * 32 + 255) / 256, 256, 0, stream>>>(gbuf, mlp_w2, mlp_b2, (float*)d_out, G);
}
